// MultiHeadFullAttention_25958782337330
// MI455X (gfx1250) — matrix operand order repaired and checked by reading, not compiled, not run on hardware
//
#include <hip/hip_runtime.h>

// ---------------------------------------------------------------------------
// MultiHeadFullAttention for MI455X (gfx1250, wave32, WMMA bf16 16x16x32)
// B=2, T=2048, D=1024, H=16, dh=64.  fp32 in/out, bf16 tensor-core math with
// fp32 accumulation, flash-attention online softmax (scores never hit HBM).
//
// Pipeline:
//   1) elementwise fp32->bf16 conversion of x_q/x_k/x_v and Wq/Wk/Wv/Wo
//   2) QKV projections: 2x2 register-blocked WMMA GEMM (32x32 per wave)
//   3) flash attention (16 q-rows per wave, 32-key steps, causal)
//   4) output projection (2x2 register-blocked WMMA GEMM, fp32 out)
// ---------------------------------------------------------------------------

typedef __attribute__((ext_vector_type(16))) __bf16 v16bf;
typedef __attribute__((ext_vector_type(8)))  __bf16 v8bf;
typedef __attribute__((ext_vector_type(4)))  __bf16 v4bf;
typedef __attribute__((ext_vector_type(8)))  float  v8f;

#define D_MODEL 1024
#define T_SEQ   2048
#define N_HEADS 16
#define D_HEAD  64
#define BATCH   2
#define NEG_BIG (-1.0e30f)

__device__ __forceinline__ v8f wmma_bf16(v16bf a, v16bf b, v8f c) {
    return __builtin_amdgcn_wmma_f32_16x16x32_bf16(
        /*neg_a=*/false, a, /*neg_b=*/false, b,
        /*c_mod=*/(short)0, c, /*reuse_a=*/false, /*reuse_b=*/false);
}

// A-matrix fragment (16x32, 16-bit): lane holds row (lane&15); lanes 0-15
// hold K=0..15, lanes 16-31 hold K=16..31 (contiguous, matching B-frags).
// koff = 8*(lane>>4); K offset used is 2*koff.
__device__ __forceinline__ v16bf a_frag_bf16(const __bf16* row, int koff) {
    v8bf lo = *(const v8bf*)(row + 2 * koff);
    v8bf hi = *(const v8bf*)(row + 2 * koff + 8);
    v16bf f;
#pragma unroll
    for (int e = 0; e < 8; ++e) { f[e] = lo[e]; f[8 + e] = hi[e]; }
    return f;
}

// B-matrix fragment (32x16, 16-bit): lane holds column (lane&15); lanes 0-15
// hold K=0..15, lanes 16-31 hold K=16..31 (contiguous).  Caller pre-adds the
// 16*(lane>>4) K offset.
__device__ __forceinline__ v16bf b_frag_bf16(const __bf16* col) {
    v8bf lo = *(const v8bf*)(col);
    v8bf hi = *(const v8bf*)(col + 8);
    v16bf f;
#pragma unroll
    for (int e = 0; e < 8; ++e) { f[e] = lo[e]; f[8 + e] = hi[e]; }
    return f;
}

// ---------------------------------------------------------------------------
// fp32 -> bf16 elementwise conversion (vectorized: float4 in, 4x bf16 out)
// ---------------------------------------------------------------------------
__global__ __launch_bounds__(256)
void cvt_f32_bf16_kernel(const float* __restrict__ src, __bf16* __restrict__ dst,
                         int n4) {
    const int i = blockIdx.x * 256 + threadIdx.x;
    if (i < n4) {
        float4 v = ((const float4*)src)[i];
        v4bf o;
        o[0] = (__bf16)v.x; o[1] = (__bf16)v.y;
        o[2] = (__bf16)v.z; o[3] = (__bf16)v.w;
        ((v4bf*)dst)[i] = o;
    }
}

// ---------------------------------------------------------------------------
// QKV projection: Y[m,n] = sum_k X[m,k] * W[n,k] + bias[n]   (nn.Linear)
// 2x2 register blocking: one wave owns a 32x32 output tile; per 32-wide
// K-step: 2 A-frags + 2 B-frags -> 4 WMMAs (1 fragment load per WMMA).
// mode 0: write bf16 as [b, h, t, dh]   (Q and K)
// mode 2: write bf16 as [b, h, dh, t]   (V transposed for PV B-frags)
// ---------------------------------------------------------------------------
__global__ __launch_bounds__(256)
void proj_qkv_kernel(const __bf16* __restrict__ X, const __bf16* __restrict__ W,
                     const float* __restrict__ bias, __bf16* __restrict__ out,
                     int mode) {
    const int wave = threadIdx.x >> 5;
    const int lane = threadIdx.x & 31;
    const int tile = blockIdx.x * 8 + wave;     // 4096 tiles (128 x 32)
    const int nt = tile & 31;                   // 32 N tiles of 32
    const int mt = tile >> 5;                   // 128 M tiles of 32
    const int m0 = mt * 32, n0 = nt * 32;

    const int l15  = lane & 15;
    const int half = lane >> 4;
    const int koff = half * 8;
    const int boff = half * 16;

    const __bf16* xrow0 = X + (size_t)(m0 + l15) * D_MODEL;
    const __bf16* xrow1 = xrow0 + (size_t)16 * D_MODEL;
    const __bf16* wcol0 = W + (size_t)(n0 + l15) * D_MODEL + boff;
    const __bf16* wcol1 = wcol0 + (size_t)16 * D_MODEL;

    v8f c[2][2] = {{v8f{}, v8f{}}, {v8f{}, v8f{}}};
#pragma unroll 2
    for (int kk = 0; kk < D_MODEL; kk += 32) {
        const v16bf a0 = a_frag_bf16(xrow0 + kk, koff);
        const v16bf a1 = a_frag_bf16(xrow1 + kk, koff);
        const v16bf b0 = b_frag_bf16(wcol0 + kk);
        const v16bf b1 = b_frag_bf16(wcol1 + kk);
        c[0][0] = wmma_bf16(a0, b0, c[0][0]);
        c[0][1] = wmma_bf16(a0, b1, c[0][1]);
        c[1][0] = wmma_bf16(a1, b0, c[1][0]);
        c[1][1] = wmma_bf16(a1, b1, c[1][1]);
    }

#pragma unroll
    for (int ni = 0; ni < 2; ++ni) {
        const int ncol = n0 + ni * 16 + l15;
        const float bn = bias[ncol];
        const int h = ncol >> 6;       // / D_HEAD
        const int d = ncol & 63;
#pragma unroll
        for (int mi = 0; mi < 2; ++mi) {
#pragma unroll
            for (int r = 0; r < 8; ++r) {
                const int mrow = m0 + mi * 16 + r + 8 * half;
                const int bI = mrow >> 11;            // / T_SEQ
                const int t  = mrow & (T_SEQ - 1);
                size_t idx;
                if (mode == 2)
                    idx = ((size_t)(bI * N_HEADS + h) * D_HEAD + d) * T_SEQ + t;
                else
                    idx = ((size_t)(bI * N_HEADS + h) * T_SEQ + t) * D_HEAD + d;
                out[idx] = (__bf16)(c[mi][ni][r] + bn);
            }
        }
    }
}

// ---------------------------------------------------------------------------
// Flash attention: one wave owns a 16-row query tile of one (b,h); streams
// keys in 32-wide steps (causal).  Per step: 4 WMMAs for S = Q K^T, online
// softmax, P transposed via per-wave LDS tile, 4 WMMAs for O += P V.
// Next step's K/V lines are prefetched (global_prefetch_b8).
// ---------------------------------------------------------------------------
__global__ __launch_bounds__(256)
void attn_kernel(const __bf16* __restrict__ Qb, const __bf16* __restrict__ Kb,
                 const __bf16* __restrict__ Vt, __bf16* __restrict__ Ao) {
    __shared__ __bf16 plds[8][16 * 32];        // one 16x32 P tile per wave

    const int wave = threadIdx.x >> 5;
    const int lane = threadIdx.x & 31;
    const int tile = blockIdx.x * 8 + wave;    // 4096 tiles = B*H*(T/16)
    const int qt = tile & 127;
    const int bh = tile >> 7;                  // b*H + h
    const int q0 = qt * 16;

    const int l15  = lane & 15;
    const int half = lane >> 4;
    const int koff = half * 8;                 // A-frag K offset
    const int boff = half * 16;                // B-frag K offset

    const __bf16* qbase = Qb + (size_t)bh * T_SEQ * D_HEAD;
    const __bf16* kbase = Kb + (size_t)bh * T_SEQ * D_HEAD;
    const __bf16* vbase = Vt + (size_t)bh * D_HEAD * T_SEQ;

    // Q fragments: rows q0+l15, two 32-wide d chunks
    const __bf16* qrow = qbase + (size_t)(q0 + l15) * D_HEAD;
    const v16bf aQ0 = a_frag_bf16(qrow, koff);
    const v16bf aQ1 = a_frag_bf16(qrow + 32, koff);

    v8f acc[4] = {v8f{}, v8f{}, v8f{}, v8f{}};
    float mrow[8], lrow[8];
#pragma unroll
    for (int r = 0; r < 8; ++r) { mrow[r] = NEG_BIG; lrow[r] = 0.f; }

    const float scale = 0.125f;                // 1/sqrt(64)
    const int ksteps = ((q0 + 15) >> 5) + 1;   // causal bound
    __bf16* myp = &plds[wave][0];

    for (int ks = 0; ks < ksteps; ++ks) {
        const int k0 = ks * 32;

        // prefetch next step's K rows / V lines into cache
        if (ks + 1 < ksteps) {
            const int kn = k0 + 32;
            __builtin_prefetch(kbase + (size_t)(kn + l15) * D_HEAD, 0, 3);
            __builtin_prefetch(kbase + (size_t)(kn + 16 + l15) * D_HEAD, 0, 3);
            __builtin_prefetch(vbase + (size_t)l15 * T_SEQ + kn, 0, 3);
        }

        // ---- S = Q K^T ----
        const __bf16* krow_lo = kbase + (size_t)(k0 + l15) * D_HEAD + boff;
        const __bf16* krow_hi = kbase + (size_t)(k0 + 16 + l15) * D_HEAD + boff;
        v8f s_lo = {}, s_hi = {};
        s_lo = wmma_bf16(aQ0, b_frag_bf16(krow_lo),      s_lo);
        s_lo = wmma_bf16(aQ1, b_frag_bf16(krow_lo + 32), s_lo);
        s_hi = wmma_bf16(aQ0, b_frag_bf16(krow_hi),      s_hi);
        s_hi = wmma_bf16(aQ1, b_frag_bf16(krow_hi + 32), s_hi);

        // ---- causal mask + scale + online softmax (rows live in groups of
        //      16 lanes; reduce across lanes with xor-shuffles 1,2,4,8) ----
        float p_lo[8], p_hi[8], alpha[8];
#pragma unroll
        for (int r = 0; r < 8; ++r) {
            const int q   = q0 + r + 8 * half;
            const int klo = k0 + l15;
            float slo = (klo > q)      ? NEG_BIG : s_lo[r] * scale;
            float shi = (klo + 16 > q) ? NEG_BIG : s_hi[r] * scale;
            float rm = fmaxf(slo, shi);
            rm = fmaxf(rm, __shfl_xor(rm, 1, 32));
            rm = fmaxf(rm, __shfl_xor(rm, 2, 32));
            rm = fmaxf(rm, __shfl_xor(rm, 4, 32));
            rm = fmaxf(rm, __shfl_xor(rm, 8, 32));
            const float mnew = fmaxf(mrow[r], rm);
            alpha[r] = __expf(mrow[r] - mnew);
            p_lo[r] = __expf(slo - mnew);
            p_hi[r] = __expf(shi - mnew);
            float rs = p_lo[r] + p_hi[r];
            rs += __shfl_xor(rs, 1, 32);
            rs += __shfl_xor(rs, 2, 32);
            rs += __shfl_xor(rs, 4, 32);
            rs += __shfl_xor(rs, 8, 32);
            lrow[r] = lrow[r] * alpha[r] + rs;
            mrow[r] = mnew;
        }
#pragma unroll
        for (int cI = 0; cI < 4; ++cI)
#pragma unroll
            for (int r = 0; r < 8; ++r) acc[cI][r] *= alpha[r];

        // ---- transpose P (C-layout) -> A-layout via per-wave LDS tile ----
#pragma unroll
        for (int r = 0; r < 8; ++r) {
            const int mr = r + 8 * half;
            myp[mr * 32 + l15]      = (__bf16)p_lo[r];
            myp[mr * 32 + 16 + l15] = (__bf16)p_hi[r];
        }
        asm volatile("s_wait_dscnt 0x0" ::: "memory");  // per-wave RAW safety
        const v16bf aP = a_frag_bf16(myp + l15 * 32, koff);

        // ---- O += P V ----
#pragma unroll
        for (int cI = 0; cI < 4; ++cI) {
            const __bf16* vcol = vbase + (size_t)(cI * 16 + l15) * T_SEQ + k0 + boff;
            acc[cI] = wmma_bf16(aP, b_frag_bf16(vcol), acc[cI]);
        }
    }

    // ---- normalize and store merged-head output [B, T, D] (bf16) ----
    const int bI = bh >> 4;
    const int h  = bh & 15;
#pragma unroll
    for (int r = 0; r < 8; ++r) {
        const int t = q0 + r + 8 * half;
        const float il = 1.0f / lrow[r];
        const size_t base = ((size_t)(bI * T_SEQ + t)) * D_MODEL + h * D_HEAD + l15;
#pragma unroll
        for (int cI = 0; cI < 4; ++cI)
            Ao[base + cI * 16] = (__bf16)(acc[cI][r] * il);
    }
}

// ---------------------------------------------------------------------------
// Output projection: Y[m,n] = sum_k Ao[m,k] * Wo[n,k] + bo[n], fp32 out.
// Same 2x2 register blocking as the QKV GEMM.
// ---------------------------------------------------------------------------
__global__ __launch_bounds__(256)
void out_proj_kernel(const __bf16* __restrict__ A, const __bf16* __restrict__ W,
                     const float* __restrict__ bias, float* __restrict__ Y) {
    const int wave = threadIdx.x >> 5;
    const int lane = threadIdx.x & 31;
    const int tile = blockIdx.x * 8 + wave;     // 4096 tiles (128 x 32)
    const int nt = tile & 31;
    const int mt = tile >> 5;
    const int m0 = mt * 32, n0 = nt * 32;

    const int l15  = lane & 15;
    const int half = lane >> 4;
    const int koff = half * 8;
    const int boff = half * 16;

    const __bf16* arow0 = A + (size_t)(m0 + l15) * D_MODEL;
    const __bf16* arow1 = arow0 + (size_t)16 * D_MODEL;
    const __bf16* wcol0 = W + (size_t)(n0 + l15) * D_MODEL + boff;
    const __bf16* wcol1 = wcol0 + (size_t)16 * D_MODEL;

    v8f c[2][2] = {{v8f{}, v8f{}}, {v8f{}, v8f{}}};
#pragma unroll 2
    for (int kk = 0; kk < D_MODEL; kk += 32) {
        const v16bf a0 = a_frag_bf16(arow0 + kk, koff);
        const v16bf a1 = a_frag_bf16(arow1 + kk, koff);
        const v16bf b0 = b_frag_bf16(wcol0 + kk);
        const v16bf b1 = b_frag_bf16(wcol1 + kk);
        c[0][0] = wmma_bf16(a0, b0, c[0][0]);
        c[0][1] = wmma_bf16(a0, b1, c[0][1]);
        c[1][0] = wmma_bf16(a1, b0, c[1][0]);
        c[1][1] = wmma_bf16(a1, b1, c[1][1]);
    }

#pragma unroll
    for (int ni = 0; ni < 2; ++ni) {
        const int ncol = n0 + ni * 16 + l15;
        const float bn = bias[ncol];
#pragma unroll
        for (int mi = 0; mi < 2; ++mi) {
#pragma unroll
            for (int r = 0; r < 8; ++r) {
                const int mrow = m0 + mi * 16 + r + 8 * half;
                Y[(size_t)mrow * D_MODEL + ncol] = c[mi][ni][r] + bn;
            }
        }
    }
}

// ---------------------------------------------------------------------------
extern "C" void kernel_launch(void* const* d_in, const int* in_sizes, int n_in,
                              void* d_out, int out_size, void* d_ws, size_t ws_size,
                              hipStream_t stream) {
    (void)in_sizes; (void)n_in; (void)out_size; (void)ws_size;
    const float* x_q = (const float*)d_in[0];
    const float* x_k = (const float*)d_in[1];
    const float* x_v = (const float*)d_in[2];
    // d_in[3] = attn_mask (causal, applied implicitly)
    const float* Wq = (const float*)d_in[4];
    const float* bq = (const float*)d_in[5];
    const float* Wk = (const float*)d_in[6];
    const float* bk = (const float*)d_in[7];
    const float* Wv = (const float*)d_in[8];
    const float* bv = (const float*)d_in[9];
    const float* Wo = (const float*)d_in[10];
    const float* bo = (const float*)d_in[11];
    // d_in[12] = n_heads (fixed at 16 here)

    const size_t XN = (size_t)BATCH * T_SEQ * D_MODEL;   // 4 Mi activations
    const size_t WN = (size_t)D_MODEL * D_MODEL;         // 1 Mi per weight

    __bf16* p = (__bf16*)d_ws;
    __bf16* Xq16 = p; p += XN;
    __bf16* Xk16 = p; p += XN;
    __bf16* Xv16 = p; p += XN;
    __bf16* Wq16 = p; p += WN;
    __bf16* Wk16 = p; p += WN;
    __bf16* Wv16 = p; p += WN;
    __bf16* Wo16 = p; p += WN;
    __bf16* Qb   = p; p += XN;   // [b, h, t, dh]
    __bf16* Kb   = p; p += XN;   // [b, h, t, dh]
    __bf16* Vt   = p; p += XN;   // [b, h, dh, t]
    __bf16* Ao   = p; p += XN;   // [b, t, D] merged-head attention output

    const dim3 blk(256);
    const int xblocks = (int)(XN / 4 / 256);   // 4096
    const int wblocks = (int)(WN / 4 / 256);   // 1024
    cvt_f32_bf16_kernel<<<xblocks, blk, 0, stream>>>(x_q, Xq16, (int)(XN / 4));
    cvt_f32_bf16_kernel<<<xblocks, blk, 0, stream>>>(x_k, Xk16, (int)(XN / 4));
    cvt_f32_bf16_kernel<<<xblocks, blk, 0, stream>>>(x_v, Xv16, (int)(XN / 4));
    cvt_f32_bf16_kernel<<<wblocks, blk, 0, stream>>>(Wq, Wq16, (int)(WN / 4));
    cvt_f32_bf16_kernel<<<wblocks, blk, 0, stream>>>(Wk, Wk16, (int)(WN / 4));
    cvt_f32_bf16_kernel<<<wblocks, blk, 0, stream>>>(Wv, Wv16, (int)(WN / 4));
    cvt_f32_bf16_kernel<<<wblocks, blk, 0, stream>>>(Wo, Wo16, (int)(WN / 4));

    proj_qkv_kernel<<<512, blk, 0, stream>>>(Xq16, Wq16, bq, Qb, 0);
    proj_qkv_kernel<<<512, blk, 0, stream>>>(Xk16, Wk16, bk, Kb, 0);
    proj_qkv_kernel<<<512, blk, 0, stream>>>(Xv16, Wv16, bv, Vt, 2);
    attn_kernel<<<512, blk, 0, stream>>>(Qb, Kb, Vt, Ao);
    out_proj_kernel<<<512, blk, 0, stream>>>(Ao, Wo16, bo, (float*)d_out);
}
